// KernelLearner_4544075399480
// MI455X (gfx1250) — matrix operand order repaired and checked by reading, not compiled, not run on hardware
//
#include <hip/hip_runtime.h>
#include <hip/hip_bf16.h>

// ---------------- problem constants (match reference) ----------------
#define B_    16
#define NC_   31
#define NX_   256
#define NY_   256
#define NO_   3
#define KXY_  17
#define NLOC_ 7533
#define KD_ELEMS (NO_*NC_*KXY_*KXY_)   // 26877 dense kernel entries
#define KH_LO 4                        // nonzero kh band is [4,12] (cy=8 +/-4)
#define KH_N  9

// A-fragment store: NO*NC*KH_N frags, each 32 lanes x 16 halves (f16)
#define AF_HALVES_PER_FRAG (32*16)                    // 512
#define AF_HALVES_PER_OC   (KH_N*AF_HALVES_PER_FRAG)  // 4608 halves = 9216 B
#define AF_BYTES_PER_OC    (AF_HALVES_PER_OC*2)       // 9216
#define AF_TOTAL_HALVES    (NO_*NC_*AF_HALVES_PER_OC) // 428544
#define AF_B128_PER_C      (3*576)                    // 1728 b128 transfers/channel

// LDS x-patch: rows x0-4 .. x0+19 (24 rows), cols y=-8..263 (272 used),
// stride padded to 288 halves (576 B) so every row is 32B aligned.
#define PATCH_ROWS   24
#define PATCH_STRIDE 288

typedef _Float16 v16h __attribute__((ext_vector_type(16)));
typedef float    v8f  __attribute__((ext_vector_type(8)));

// ---------------- setup kernels ----------------
__global__ void k_zero_kd(float* kd) {
  int i = blockIdx.x * 256 + threadIdx.x;
  if (i < KD_ELEMS) kd[i] = 0.f;
}

__global__ void k_scatter(const float* __restrict__ vk,
                          const int* __restrict__ loc,
                          float* __restrict__ kd) {
  int i = blockIdx.x * 256 + threadIdx.x;
  if (i < NLOC_) {
    float v = vk[i];
    kd[loc[i]] = v > 0.f ? v : 0.f;   // ReLU; locations are unique
  }
}

// Build banded Toeplitz A fragments in contiguous k order (same as the B loads):
// element (lane l, half h) holds A[m=l&15, K = h + 16*(l>>4)],
// A[m,K] = kernel[o,c,kh,K-m] if 0 <= K-m <= 16 else 0.
__global__ void k_build_afrag(const float* __restrict__ kd,
                              _Float16* __restrict__ af) {
  int t = blockIdx.x * 256 + threadIdx.x;
  if (t >= AF_TOTAL_HALVES) return;
  int h    = t & 15;
  int l    = (t >> 4) & 31;
  int rest = t >> 9;
  int kh4  = rest % KH_N;  rest /= KH_N;
  int c    = rest % NC_;
  int o    = rest / NC_;
  int m  = l & 15;
  int K  = h + 16 * (l >> 4);
  int kw = K - m;
  float v = 0.f;
  if (kw >= 0 && kw <= 16)
    v = kd[((o * NC_ + c) * KXY_ + (kh4 + KH_LO)) * KXY_ + kw];
  af[t] = (_Float16)v;
}

// ---------------- main conv kernel ----------------
// grid = (16 x-tiles, 16 batches), block = 256 threads = 8 waves.
// wave w handles TWO 16x16 output tiles (y0 = 32w and 32w+16) for all 3
// orders: 6 accumulators reuse each A fragment across 2 B fragments.
// A fragments are double-buffered in LDS and staged with ASYNCcnt-tracked
// global_load_async_to_lds_b128, overlapping the next channel's fetch with
// this channel's WMMA work.
__global__ __launch_bounds__(256)
void k_conv(const float* __restrict__ x,
            const _Float16* __restrict__ af,
            float* __restrict__ out) {
  __shared__ __align__(64) _Float16 s_patch[PATCH_ROWS * PATCH_STRIDE]; // 13824 B
  __shared__ __align__(64) _Float16 s_af[2][NO_ * AF_HALVES_PER_OC];    // 55296 B

  const int tid  = threadIdx.x;
  const int lane = tid & 31;
  const int wave = tid >> 5;
  const int x0   = blockIdx.x * 16;
  const int b    = blockIdx.y;
  const int y0   = wave * 32;   // this wave's first y-tile; second is y0+16
  const int n    = lane & 15;   // N index (output x within tile)
  const int lhi  = lane >> 4;

  // LDS byte offset: shared aperture occupies the high 32 bits of the flat
  // address, so truncation yields the DS byte address for async-to-LDS.
  const unsigned af_lds = (unsigned)(uintptr_t)(void*)&s_af[0][0];

  // Issue one channel's A fragments (1728 b128) into LDS buffer `buf`.
  // Padded to 7 issues per thread (1792) so every wave's ASYNCcnt rises by
  // exactly 7; the 64 duplicate transfers rewrite identical bytes (benign).
  auto issue_af = [&](int c, int buf) {
#pragma unroll
    for (int k = 0; k < 7; ++k) {
      int i   = tid + k * 256;
      int idx = (i < AF_B128_PER_C) ? i : (i - 256);
      int o = idx / 576, j = idx - o * 576;
      const char* saddr = (const char*)af + (size_t)c * AF_BYTES_PER_OC;
      unsigned voff = (unsigned)(o * (NC_ * AF_BYTES_PER_OC) + j * 16);
      unsigned dst  = af_lds + (unsigned)(buf * (NO_ * AF_BYTES_PER_OC) + idx * 16);
      asm volatile("global_load_async_to_lds_b128 %0, %1, %2"
                   :: "v"(dst), "v"(voff), "s"(saddr)
                   : "memory");
    }
  };

  // One-time zero fill of the always-zero y-pad borders (cols 0-7, 264-287).
  for (int i = tid; i < PATCH_ROWS * 32; i += 256) {  // 3 iters, exact
    int r = i >> 5, j = i & 31;
    int col = (j < 8) ? j : (256 + j);
    s_patch[r * PATCH_STRIDE + col] = (_Float16)0.f;
  }

  issue_af(0, 0);  // preload channel 0

  v8f acc00 = {}, acc01 = {};   // o=0, y-tiles 0/1
  v8f acc10 = {}, acc11 = {};   // o=1
  v8f acc20 = {}, acc21 = {};   // o=2

  for (int c = 0; c < NC_; ++c) {
    __syncthreads();  // previous iteration's LDS reads done

    // --- stage x patch interior (float4 global -> 4x f16 LDS b64 stores) ---
    const float* xc = x + (size_t)(b * NC_ + c) * (NX_ * NY_);
#pragma unroll
    for (int k = 0; k < 6; ++k) {          // 24 rows x 64 float4 = 1536 = 6*256
      int i = tid + k * 256;
      int r = i >> 6, q = i & 63;
      int gx = x0 - 4 + r;                 // input row
      float4 f = make_float4(0.f, 0.f, 0.f, 0.f);
      if ((unsigned)gx < (unsigned)NX_)
        f = *(const float4*)(xc + gx * NY_ + q * 4);
      union { _Float16 h[4]; uint2 u2; } p;
      p.h[0] = (_Float16)f.x; p.h[1] = (_Float16)f.y;
      p.h[2] = (_Float16)f.z; p.h[3] = (_Float16)f.w;
      *(uint2*)(s_patch + r * PATCH_STRIDE + 8 + q * 4) = p.u2;
    }

    // --- double-buffered async A staging: issue c+1, wait for batch c ---
    if (c + 1 < NC_) {
      issue_af(c + 1, (c + 1) & 1);
      // 7 transfers of batch c+1 outstanding -> waiting <=7 completes batch c
      asm volatile("s_wait_asynccnt 0x7" ::: "memory");
      // prefetch next channel's patch rows into cache
      if (tid < PATCH_ROWS) {
        int gx = x0 - 4 + tid;
        if ((unsigned)gx < (unsigned)NX_)
          __builtin_prefetch(x + (size_t)(b * NC_ + c + 1) * (NX_ * NY_) + gx * NY_, 0, 1);
      }
    } else {
      asm volatile("s_wait_asynccnt 0x0" ::: "memory");
    }
    __syncthreads();

    // --- 9 kh taps x 3 orders x 2 y-tiles of v_wmma_f32_16x16x32_f16 ---
    const _Float16* afb = &s_af[c & 1][0];
#pragma unroll
    for (int kh4 = 0; kh4 < KH_N; ++kh4) {
      // B[k,n] = patch[row = n + kh - 4, col = ytile_y0 + k]; contiguous k order (as A):
      // lane group lhi holds K = 16*lhi + h -> one contiguous 32B run per lane.
      const int rrow = n + kh4;  // n + (kh4+4) - 4
      const _Float16* prow = s_patch + rrow * PATCH_STRIDE + y0 + 16 * lhi;
      const v16h b0 = *(const v16h*)(prow);
      const v16h b1 = *(const v16h*)(prow + 16);
      const v16h a0 = *(const v16h*)(afb + ((0 * KH_N + kh4) * 32 + lane) * 16);
      const v16h a1 = *(const v16h*)(afb + ((1 * KH_N + kh4) * 32 + lane) * 16);
      const v16h a2 = *(const v16h*)(afb + ((2 * KH_N + kh4) * 32 + lane) * 16);
      acc00 = __builtin_amdgcn_wmma_f32_16x16x32_f16(false, a0, false, b0,
                                                     (short)0, acc00, false, false);
      acc01 = __builtin_amdgcn_wmma_f32_16x16x32_f16(false, a0, false, b1,
                                                     (short)0, acc01, false, false);
      acc10 = __builtin_amdgcn_wmma_f32_16x16x32_f16(false, a1, false, b0,
                                                     (short)0, acc10, false, false);
      acc11 = __builtin_amdgcn_wmma_f32_16x16x32_f16(false, a1, false, b1,
                                                     (short)0, acc11, false, false);
      acc20 = __builtin_amdgcn_wmma_f32_16x16x32_f16(false, a2, false, b0,
                                                     (short)0, acc20, false, false);
      acc21 = __builtin_amdgcn_wmma_f32_16x16x32_f16(false, a2, false, b1,
                                                     (short)0, acc21, false, false);
    }
  }

  // --- writeback: C/D layout -> lane holds 8 consecutive y at fixed x ---
  // VGPR r, lane l -> (M = r + 8*lhi, N = l&15); y = ytile_y0 + M, x = x0 + N.
  union { v8f v; float4 q[2]; } u;
#define WRITE_TILE(ACC, O, T)                                                  \
  {                                                                            \
    u.v = ACC;                                                                 \
    size_t base = ((size_t)(b * NO_ + (O)) * NX_ + (x0 + n)) * NY_             \
                  + y0 + 16 * (T) + 8 * lhi;                                   \
    *(float4*)(out + base)     = u.q[0];                                       \
    *(float4*)(out + base + 4) = u.q[1];                                       \
  }
  WRITE_TILE(acc00, 0, 0) WRITE_TILE(acc01, 0, 1)
  WRITE_TILE(acc10, 1, 0) WRITE_TILE(acc11, 1, 1)
  WRITE_TILE(acc20, 2, 0) WRITE_TILE(acc21, 2, 1)
#undef WRITE_TILE
}

// ---------------- launcher ----------------
extern "C" void kernel_launch(void* const* d_in, const int* in_sizes, int n_in,
                              void* d_out, int out_size, void* d_ws, size_t ws_size,
                              hipStream_t stream) {
  const float* x   = (const float*)d_in[0];
  const float* vk  = (const float*)d_in[1];
  const int*   loc = (const int*)d_in[2];
  float* out = (float*)d_out;

  // workspace layout: [0, 107520) dense kernel f32; then A fragments f16
  char* ws = (char*)d_ws;
  float*    kd = (float*)ws;
  _Float16* af = (_Float16*)(ws + 107520);  // 26877*4 rounded to 256

  k_zero_kd<<<(KD_ELEMS + 255) / 256, 256, 0, stream>>>(kd);
  k_scatter<<<(NLOC_ + 255) / 256, 256, 0, stream>>>(vk, loc, kd);
  k_build_afrag<<<(AF_TOTAL_HALVES + 255) / 256, 256, 0, stream>>>(kd, af);

  dim3 grid(NX_ / 16, B_);   // (16 x-tiles, 16 batches)
  k_conv<<<grid, 256, 0, stream>>>(x, af, out);
}
